// TransformerBlock_86595130622541
// MI455X (gfx1250) — matrix operand order repaired and checked by reading, not compiled, not run on hardware
//
#include <hip/hip_runtime.h>
#include <stdint.h>

#define DIM     1024
#define NHEAD   16
#define HDIM    64
#define HIDDEN  4096
#define BATCH   2
#define SEQ     2048
#define TOKENS  (BATCH * SEQ)

typedef __attribute__((ext_vector_type(16))) __bf16 v16bf;
typedef __attribute__((ext_vector_type(8)))  float  v8f;

struct __align__(16) U4 { unsigned x, y, z, w; };

union Frag {
    v16bf v;
    U4 q[2];
    unsigned short us[16];
};

__device__ __forceinline__ unsigned short f2bf(float f) {
    union { float f; unsigned u; } c; c.f = f;
    unsigned r = (c.u + 0x7FFFu + ((c.u >> 16) & 1u)) >> 16;
    return (unsigned short)r;
}

__device__ __forceinline__ v8f wmma_bf16(v16bf a, v16bf b, v8f c) {
    // v_wmma_f32_16x16x32_bf16: D = A(16x32) * B(32x16) + C(16x16 f32)
    return __builtin_amdgcn_wmma_f32_16x16x32_bf16(
        /*neg_a=*/false, a, /*neg_b=*/false, b,
        /*c_mod=*/(short)0, c, /*reuse_a=*/false, /*reuse_b=*/false);
}

__device__ __forceinline__ v8f v8f_zero() {
    v8f z;
#pragma unroll
    for (int e = 0; e < 8; ++e) z[e] = 0.0f;
    return z;
}

// ---------------------------------------------------------------------------
// Transpose + fp32 -> bf16 convert:  src [R][C] f32  ->  dst [C][R] bf16
// ---------------------------------------------------------------------------
__global__ void tcvt_kernel(const float* __restrict__ src,
                            unsigned short* __restrict__ dst,
                            int R, int C) {
    int idx = blockIdx.x * 256 + threadIdx.x;
    if (idx >= R * C) return;
    int r = idx / C;
    int c = idx - r * C;
    dst[(size_t)c * R + r] = f2bf(src[idx]);
}

// ---------------------------------------------------------------------------
// RMSNorm: one block (256 threads) per token row of 1024 f32 -> bf16
// ---------------------------------------------------------------------------
__global__ __launch_bounds__(256) void rmsnorm_kernel(
    const float* __restrict__ x, const float* __restrict__ g,
    unsigned short* __restrict__ out) {
    int row = blockIdx.x;
    const float* xr = x + (size_t)row * DIM;
    int tid = threadIdx.x;
    float v[4];
    float ss = 0.f;
#pragma unroll
    for (int i = 0; i < 4; ++i) {
        v[i] = xr[tid + i * 256];
        ss += v[i] * v[i];
    }
#pragma unroll
    for (int off = 16; off; off >>= 1) ss += __shfl_xor(ss, off, 32);
    __shared__ float red[8];
    int wave = tid >> 5, lane = tid & 31;
    if (lane == 0) red[wave] = ss;
    __syncthreads();
    float tot = 0.f;
#pragma unroll
    for (int i = 0; i < 8; ++i) tot += red[i];
    float norm = rsqrtf(tot / (float)DIM + 1e-6f);
    unsigned short* orow = out + (size_t)row * DIM;
#pragma unroll
    for (int i = 0; i < 4; ++i) {
        int c = tid + i * 256;
        orow[c] = f2bf(v[i] * norm * g[c]);
    }
}

// ---------------------------------------------------------------------------
// WMMA GEMM:  C[M][N] = A[M][K](bf16, row-major) * Bt[N][K](bf16)^T
// Block tile 128x128, 8 waves (4 in M, 2 in N), wave tile 32x64.
// Epilogue modes fuse output layout / activation / residual.
// ---------------------------------------------------------------------------
constexpr int MODE_QK    = 0;  // bf16 out, [B,H,T,D] per-head layout
constexpr int MODE_VT    = 1;  // bf16 out, [B,H,D,T] transposed layout
constexpr int MODE_RESID = 2;  // f32 out = acc + resid[m][n]
constexpr int MODE_SILU  = 3;  // bf16 out = silu(acc), [M][N]

template <int MODE>
__global__ __launch_bounds__(256) void gemm_kernel(
    const unsigned short* __restrict__ A,
    const unsigned short* __restrict__ Bt,
    const float* __restrict__ resid,
    void* __restrict__ out,
    int M, int N, int K) {
    const int lane = threadIdx.x & 31;
    const int wave = threadIdx.x >> 5;
    const int l16  = lane & 15;
    const int half = lane >> 4;
    const int kb   = half * 8;

    const int bm = blockIdx.y * 128;
    const int bn = blockIdx.x * 128;
    const int wm = (wave & 3) * 32;
    const int wn = (wave >> 2) * 64;

    v8f acc[2][4];
#pragma unroll
    for (int i = 0; i < 2; ++i)
#pragma unroll
        for (int j = 0; j < 4; ++j) acc[i][j] = v8f_zero();

    const unsigned short* Arow[2];
    Arow[0] = A + (size_t)(bm + wm + l16) * K;
    Arow[1] = A + (size_t)(bm + wm + 16 + l16) * K;
    const unsigned short* Brow[4];
#pragma unroll
    for (int j = 0; j < 4; ++j)
        Brow[j] = Bt + (size_t)(bn + wn + j * 16 + l16) * K;

    for (int k0 = 0; k0 < K; k0 += 32) {
        Frag a[2], b[4];
#pragma unroll
        for (int i = 0; i < 2; ++i) {
            a[i].q[0] = *(const U4*)(Arow[i] + k0 + half * 16);
            a[i].q[1] = *(const U4*)(Arow[i] + k0 + half * 16 + 8);
        }
#pragma unroll
        for (int j = 0; j < 4; ++j) {
            const U4* p = (const U4*)(Brow[j] + k0 + half * 16);
            b[j].q[0] = p[0];
            b[j].q[1] = p[1];
        }
        __builtin_prefetch(Arow[0] + k0 + 128, 0, 1);
        __builtin_prefetch(Brow[0] + k0 + 128, 0, 1);
        __builtin_prefetch(Brow[2] + k0 + 128, 0, 1);
#pragma unroll
        for (int i = 0; i < 2; ++i)
#pragma unroll
            for (int j = 0; j < 4; ++j)
                acc[i][j] = wmma_bf16(a[i].v, b[j].v, acc[i][j]);
    }

#pragma unroll
    for (int i = 0; i < 2; ++i) {
#pragma unroll
        for (int j = 0; j < 4; ++j) {
#pragma unroll
            for (int r = 0; r < 8; ++r) {
                int m = bm + wm + i * 16 + r + half * 8;
                int n = bn + wn + j * 16 + l16;
                float val = acc[i][j][r];
                if constexpr (MODE == MODE_QK) {
                    int b_ = m >> 11, t = m & 2047, h = n >> 6, d = n & 63;
                    ((unsigned short*)out)[(((size_t)(b_ * NHEAD + h) * SEQ + t) << 6) + d] = f2bf(val);
                } else if constexpr (MODE == MODE_VT) {
                    int b_ = m >> 11, t = m & 2047, h = n >> 6, d = n & 63;
                    ((unsigned short*)out)[((size_t)(b_ * NHEAD + h) * HDIM + d) * SEQ + t] = f2bf(val);
                } else if constexpr (MODE == MODE_RESID) {
                    size_t idx = (size_t)m * N + n;
                    ((float*)out)[idx] = val + resid[idx];
                } else {  // MODE_SILU
                    float s = val / (1.0f + __expf(-val));
                    ((unsigned short*)out)[(size_t)m * N + n] = f2bf(s);
                }
            }
        }
    }
}

// ---------------------------------------------------------------------------
// Flash attention: 4 waves/block, one 16-query tile per wave, 32 keys/step.
// S = Q*K^T via 4 WMMAs, online softmax (max via width-16 shfl only),
// softmax denominator accumulated by an extra "ones-column" WMMA,
// P staged via LDS (C-layout -> A-layout), P*V via 4 full-K WMMAs vs V^T.
// ---------------------------------------------------------------------------
__global__ __launch_bounds__(128) void attn_kernel(
    const unsigned short* __restrict__ qh,   // [B,H,T,D] bf16
    const unsigned short* __restrict__ kh,   // [B,H,T,D] bf16
    const unsigned short* __restrict__ vt,   // [B,H,D,T] bf16
    unsigned short* __restrict__ out) {      // [B*T, DIM] bf16
    const int lane = threadIdx.x & 31;
    const int wave = threadIdx.x >> 5;
    const int l16  = lane & 15;
    const int half = lane >> 4;
    const int kb   = half * 8;

    const int qt = blockIdx.x * 4 + wave;
    const int h = blockIdx.y, b = blockIdx.z;
    const int bh = b * NHEAD + h;
    const int qrow_base = qt * 16;

    const unsigned short* Q  = qh + ((size_t)bh * SEQ + qrow_base) * HDIM;
    const unsigned short* Kp = kh + (size_t)bh * SEQ * HDIM;
    const unsigned short* Vp = vt + (size_t)bh * HDIM * SEQ;

    // Q fragments (two K-chunks of 32 over head-dim 64), A layout
    Frag qf[2];
#pragma unroll
    for (int c = 0; c < 2; ++c) {
        qf[c].q[0] = *(const U4*)(Q + l16 * HDIM + c * 32 + half * 16);
        qf[c].q[1] = *(const U4*)(Q + l16 * HDIM + c * 32 + half * 16 + 8);
    }

    // "ones in column 0" B fragment: o[4] column 0 accumulates row-sum(P)
    Frag onef;
    {
        unsigned fill = (l16 == 0) ? 0x3F803F80u : 0u;  // two bf16 1.0
        onef.q[0].x = onef.q[0].y = onef.q[0].z = onef.q[0].w = fill;
        onef.q[1].x = onef.q[1].y = onef.q[1].z = onef.q[1].w = fill;
    }

    v8f o[5];  // 4 d-tiles + softmax-denominator accumulator
#pragma unroll
    for (int t = 0; t < 5; ++t) o[t] = v8f_zero();
    float mrow[8];
#pragma unroll
    for (int r = 0; r < 8; ++r) mrow[r] = -1e30f;

    __shared__ __align__(16) unsigned short pls[4][16 * 32];
    unsigned short* P = pls[wave];

    const int nsteps = qrow_base / 32 + 1;   // covers keys [0, n*32) <= 2048
    for (int it = 0; it < nsteps; ++it) {
        const int s0 = it * 32;
        // ---- two S tiles = Q * K^T over keys [s0, s0+32) ----
        v8f sa = v8f_zero(), sb = v8f_zero();
#pragma unroll
        for (int c = 0; c < 2; ++c) {
            Frag kfa, kfb;
            const U4* pa_ = (const U4*)(Kp + (size_t)(s0 + l16) * HDIM + c * 32 + half * 16);
            const U4* pb_ = (const U4*)(Kp + (size_t)(s0 + 16 + l16) * HDIM + c * 32 + half * 16);
            kfa.q[0] = pa_[0]; kfa.q[1] = pa_[1];
            kfb.q[0] = pb_[0]; kfb.q[1] = pb_[1];
            sa = wmma_bf16(qf[c].v, kfa.v, sa);
            sb = wmma_bf16(qf[c].v, kfb.v, sb);
        }
        // ---- scale, causal mask, online softmax (max-only reductions) ----
        float pra[8], prb[8];
#pragma unroll
        for (int r = 0; r < 8; ++r) {
            const int row = qrow_base + r + half * 8;
            float va = sa[r] * 0.125f;                   // 1/sqrt(64)
            float vb = sb[r] * 0.125f;
            if (s0 + l16 > row)      va = -1e9f;
            if (s0 + 16 + l16 > row) vb = -1e9f;
            float mx = fmaxf(va, vb);
            mx = fmaxf(mx, __shfl_xor(mx, 1, 16));
            mx = fmaxf(mx, __shfl_xor(mx, 2, 16));
            mx = fmaxf(mx, __shfl_xor(mx, 4, 16));
            mx = fmaxf(mx, __shfl_xor(mx, 8, 16));
            float nm = fmaxf(mrow[r], mx);
            float alpha = __expf(mrow[r] - nm);
            mrow[r] = nm;
            pra[r] = __expf(va - nm);
            prb[r] = __expf(vb - nm);
            o[0][r] *= alpha;
            o[1][r] *= alpha;
            o[2][r] *= alpha;
            o[3][r] *= alpha;
            o[4][r] *= alpha;   // denominator rescales identically
        }
        // ---- C-layout -> A-layout via LDS (per-wave tile; DS in-order) ----
#pragma unroll
        for (int r = 0; r < 8; ++r) {
            const int row = r + half * 8;
            P[row * 32 + l16]      = f2bf(pra[r]);
            P[row * 32 + 16 + l16] = f2bf(prb[r]);
        }
        asm volatile("s_wait_dscnt 0x0" ::: "memory");
        Frag pfr;
        pfr.q[0] = *(const U4*)(&P[l16 * 32 + half * 16]);
        pfr.q[1] = *(const U4*)(&P[l16 * 32 + half * 16 + 8]);
        // ---- O += P * V  (full K=32 contraction, contiguous V^T loads) ----
#pragma unroll
        for (int dt = 0; dt < 4; ++dt) {
            Frag vf;
            const U4* vp = (const U4*)(Vp + (size_t)(dt * 16 + l16) * SEQ + s0 + half * 16);
            vf.q[0] = vp[0];
            vf.q[1] = vp[1];
            o[dt] = wmma_bf16(pfr.v, vf.v, o[dt]);
        }
        o[4] = wmma_bf16(pfr.v, onef.v, o[4]);   // row-sum accumulation
    }

    // ---- normalize (broadcast denominator from column-0 lanes) and store ----
    float lrow[8];
#pragma unroll
    for (int r = 0; r < 8; ++r) lrow[r] = __shfl(o[4][r], 0, 16);
#pragma unroll
    for (int dt = 0; dt < 4; ++dt) {
#pragma unroll
        for (int r = 0; r < 8; ++r) {
            float val = o[dt][r] / lrow[r];
            int t = qrow_base + r + half * 8;
            out[((size_t)(b * SEQ + t)) * DIM + h * HDIM + dt * 16 + l16] = f2bf(val);
        }
    }
}

// ---------------------------------------------------------------------------
// Host launcher
// ---------------------------------------------------------------------------
extern "C" void kernel_launch(void* const* d_in, const int* in_sizes, int n_in,
                              void* d_out, int out_size, void* d_ws, size_t ws_size,
                              hipStream_t stream) {
    const float* x      = (const float*)d_in[0];
    // d_in[1] = mask (causal mask applied analytically in-kernel)
    const float* wq     = (const float*)d_in[2];
    const float* wk     = (const float*)d_in[3];
    const float* wv     = (const float*)d_in[4];
    const float* wo     = (const float*)d_in[5];
    const float* w1     = (const float*)d_in[6];
    const float* w2     = (const float*)d_in[7];
    const float* g_attn = (const float*)d_in[8];
    const float* g_ffn  = (const float*)d_in[9];

    char* ws = (char*)d_ws;
    size_t off = 0;
    auto alloc = [&](size_t bytes) -> char* {
        char* p = ws + off;
        off += (bytes + 255) & ~(size_t)255;
        return p;
    };

    unsigned short* wqT = (unsigned short*)alloc((size_t)DIM * DIM * 2);
    unsigned short* wkT = (unsigned short*)alloc((size_t)DIM * DIM * 2);
    unsigned short* wvT = (unsigned short*)alloc((size_t)DIM * DIM * 2);
    unsigned short* woT = (unsigned short*)alloc((size_t)DIM * DIM * 2);
    unsigned short* w1T = (unsigned short*)alloc((size_t)DIM * HIDDEN * 2);
    unsigned short* w2T = (unsigned short*)alloc((size_t)HIDDEN * DIM * 2);
    unsigned short* xn  = (unsigned short*)alloc((size_t)TOKENS * DIM * 2);
    unsigned short* qh  = (unsigned short*)alloc((size_t)TOKENS * DIM * 2);
    unsigned short* kh  = (unsigned short*)alloc((size_t)TOKENS * DIM * 2);
    unsigned short* vtb = (unsigned short*)alloc((size_t)TOKENS * DIM * 2);
    unsigned short* ao  = (unsigned short*)alloc((size_t)TOKENS * DIM * 2);
    float*          x1  = (float*)alloc((size_t)TOKENS * DIM * 4);
    unsigned short* h2  = (unsigned short*)alloc((size_t)TOKENS * DIM * 2);
    unsigned short* mid = (unsigned short*)alloc((size_t)TOKENS * HIDDEN * 2);

    // 1) weight transpose + bf16 convert (Bt = [N][K])
    tcvt_kernel<<<(DIM * DIM + 255) / 256, 256, 0, stream>>>(wq, wqT, DIM, DIM);
    tcvt_kernel<<<(DIM * DIM + 255) / 256, 256, 0, stream>>>(wk, wkT, DIM, DIM);
    tcvt_kernel<<<(DIM * DIM + 255) / 256, 256, 0, stream>>>(wv, wvT, DIM, DIM);
    tcvt_kernel<<<(DIM * DIM + 255) / 256, 256, 0, stream>>>(wo, woT, DIM, DIM);
    tcvt_kernel<<<(DIM * HIDDEN + 255) / 256, 256, 0, stream>>>(w1, w1T, DIM, HIDDEN);
    tcvt_kernel<<<(HIDDEN * DIM + 255) / 256, 256, 0, stream>>>(w2, w2T, HIDDEN, DIM);

    // 2) pre-attention RMSNorm
    rmsnorm_kernel<<<TOKENS, 256, 0, stream>>>(x, g_attn, xn);

    // 3) QKV projections (fused per-head / transposed-V layouts)
    dim3 g1(DIM / 128, TOKENS / 128);
    gemm_kernel<MODE_QK><<<g1, 256, 0, stream>>>(xn, wqT, nullptr, qh, TOKENS, DIM, DIM);
    gemm_kernel<MODE_QK><<<g1, 256, 0, stream>>>(xn, wkT, nullptr, kh, TOKENS, DIM, DIM);
    gemm_kernel<MODE_VT><<<g1, 256, 0, stream>>>(xn, wvT, nullptr, vtb, TOKENS, DIM, DIM);

    // 4) flash attention (4 query tiles per block)
    dim3 ga(SEQ / 64, NHEAD, BATCH);
    attn_kernel<<<ga, 128, 0, stream>>>(qh, kh, vtb, ao);

    // 5) output projection + residual  -> x1 (f32)
    gemm_kernel<MODE_RESID><<<g1, 256, 0, stream>>>(ao, woT, x, x1, TOKENS, DIM, DIM);

    // 6) pre-FFN RMSNorm
    rmsnorm_kernel<<<TOKENS, 256, 0, stream>>>(x1, g_ffn, h2);

    // 7) FFN up + SiLU
    dim3 g2(HIDDEN / 128, TOKENS / 128);
    gemm_kernel<MODE_SILU><<<g2, 256, 0, stream>>>(h2, w1T, nullptr, mid, TOKENS, HIDDEN, DIM);

    // 8) FFN down + residual -> d_out (f32)
    gemm_kernel<MODE_RESID><<<g1, 256, 0, stream>>>(mid, w2T, x1, d_out, TOKENS, DIM, HIDDEN);
}
